// Decoder_1511828488364
// MI455X (gfx1250) — matrix operand order repaired and checked by reading, not compiled, not run on hardware
//
#include <hip/hip_runtime.h>

typedef _Float16 h16;
typedef __attribute__((ext_vector_type(16))) _Float16 v16h;
typedef __attribute__((ext_vector_type(8)))  _Float16 v8h;
typedef __attribute__((ext_vector_type(8)))  float    v8f;

#define B_   4
#define C_   96
#define H_   128
#define W_   256
#define HW_  (H_*W_)
#define CT_  471          // total feat channels
#define NEGV 0.1f

static __device__ __forceinline__ float leakyf(float v) { return v >= 0.f ? v : NEGV * v; }

// ---------------------------------------------------------------------------
// Weight repack: w (Cout,Cin,3,3) f32 -> f16 [9][CoutP][CinP], zero padded.
// ---------------------------------------------------------------------------
__global__ void repack_kernel(const float* __restrict__ w, h16* __restrict__ out,
                              int cout, int cin, int coutP, int cinP) {
  int idx = blockIdx.x * 256 + threadIdx.x;
  int total = 9 * coutP * cinP;
  if (idx >= total) return;
  int ci   = idx % cinP;
  int rest = idx / cinP;
  int co   = rest % coutP;
  int tap  = rest / coutP;
  h16 v = (h16)0.f;
  if (co < cout && ci < cin) v = (h16)w[((size_t)co * cin + ci) * 9 + tap];
  out[idx] = v;
}

// ---------------------------------------------------------------------------
// ConvTranspose 4x4 stride2 pad1: in (B,cin,64,128) -> 2 channels at outCh0.
// out[oy,ox] = sum_{ky,kx valid} in[iy,ix] * w[ci][co][ky][kx], iy=(oy+1-ky)/2
// ---------------------------------------------------------------------------
__global__ void upconv_kernel(const float* __restrict__ in, const float* __restrict__ w,
                              const float* __restrict__ bias, int cin, int outCh0,
                              float* __restrict__ featF32, h16* __restrict__ feat16) {
  int idx = blockIdx.x * 256 + threadIdx.x;
  if (idx >= B_ * HW_) return;
  int ox = idx & (W_ - 1);
  int oy = (idx >> 8) & (H_ - 1);
  int b  = idx >> 15;
  float a0 = bias[0], a1 = bias[1];
  int ky0 = (oy + 1) & 1;
  int kx0 = (ox + 1) & 1;
  for (int ky = ky0; ky < 4; ky += 2) {
    int iy = (oy + 1 - ky) >> 1;
    if (iy < 0 || iy >= H_/2) continue;
    for (int kx = kx0; kx < 4; kx += 2) {
      int ix = (ox + 1 - kx) >> 1;
      if (ix < 0 || ix >= W_/2) continue;
      const float* ip = in + (size_t)b * cin * (HW_/4) + iy * (W_/2) + ix;
      const float* wp = w + ky * 4 + kx;
      for (int ci = 0; ci < cin; ++ci) {
        float v = ip[(size_t)ci * (HW_/4)];
        a0 += v * wp[ci * 32];        // w[((ci*2+0)*4+ky)*4+kx]
        a1 += v * wp[ci * 32 + 16];   // w[((ci*2+1)*4+ky)*4+kx]
      }
    }
  }
  size_t o = (((size_t)b * CT_ + outCh0) * H_ + oy) * W_ + ox;
  featF32[o]       = a0;  feat16[o]       = (h16)a0;
  featF32[o + HW_] = a1;  feat16[o + HW_] = (h16)a1;
}

// ---------------------------------------------------------------------------
// Rotation flow, evaluated directly at the bilinear resize sample points.
// resize 512x1024 -> 128x256 samples src (4y+1.5, 4x+1.5) = avg of 4 pixels.
// Output = mean(4 evals) / INT_RESIZE, written to feat ch 369,370.
// ---------------------------------------------------------------------------
__global__ void rflow_kernel(const float* __restrict__ quat, const float* __restrict__ Km,
                             const float* __restrict__ Kinv,
                             float* __restrict__ featF32, h16* __restrict__ feat16) {
  int idx = blockIdx.x * 256 + threadIdx.x;
  if (idx >= B_ * HW_) return;
  int x = idx & (W_ - 1);
  int y = (idx >> 8) & (H_ - 1);
  int b = idx >> 15;
  const float* q = quat + b * 4;
  float r = q[0], i = q[1], j = q[2], k = q[3];
  float s = 2.f / (r*r + i*i + j*j + k*k);
  float R[9] = {1.f - s*(j*j + k*k), s*(i*j - k*r),       s*(i*k + j*r),
                s*(i*j + k*r),       1.f - s*(i*i + k*k), s*(j*k - i*r),
                s*(i*k - j*r),       s*(j*k + i*r),       1.f - s*(i*i + j*j)};
  const float* K  = Km + b * 9;
  const float* Ki = Kinv + b * 9;
  float T[9], M[9];
  #pragma unroll
  for (int rr = 0; rr < 3; ++rr)
    #pragma unroll
    for (int cc = 0; cc < 3; ++cc)
      T[rr*3+cc] = K[rr*3+0]*R[0*3+cc] + K[rr*3+1]*R[1*3+cc] + K[rr*3+2]*R[2*3+cc];
  #pragma unroll
  for (int rr = 0; rr < 3; ++rr)
    #pragma unroll
    for (int cc = 0; cc < 3; ++cc)
      M[rr*3+cc] = T[rr*3+0]*Ki[0*3+cc] + T[rr*3+1]*Ki[1*3+cc] + T[rr*3+2]*Ki[2*3+cc];
  float su = 0.f, sv = 0.f;
  #pragma unroll
  for (int dy = 1; dy <= 2; ++dy)
    #pragma unroll
    for (int dx = 1; dx <= 2; ++dx) {
      float px = 4.f * x + dx, py = 4.f * y + dy;
      float inv = 1.f / (M[6]*px + M[7]*py + M[8]);
      su += (M[0]*px + M[1]*py + M[2]) * inv - px;
      sv += (M[3]*px + M[4]*py + M[5]) * inv - py;
    }
  float u = su * 0.0625f;   // *0.25 (mean) * 0.25 (/INT_RESIZE)
  float v = sv * 0.0625f;
  size_t o = (((size_t)b * CT_ + 369) * H_ + y) * W_ + x;
  featF32[o]       = u;  feat16[o]       = (h16)u;
  featF32[o + HW_] = v;  feat16[o + HW_] = (h16)v;
}

// ---------------------------------------------------------------------------
// Copy tenOne into feat channels 371..466 (f32 + f16 mirror).
// ---------------------------------------------------------------------------
__global__ void copy_tenone_kernel(const float* __restrict__ src,
                                   float* __restrict__ featF32, h16* __restrict__ feat16) {
  int idx = blockIdx.x * 256 + threadIdx.x;
  if (idx >= B_ * C_ * HW_) return;
  int b  = idx / (C_ * HW_);
  int rr = idx % (C_ * HW_);
  int c  = rr >> 15;
  int xy = rr & (HW_ - 1);
  float v = src[idx];
  size_t o = ((size_t)b * CT_ + 371 + c) * HW_ + xy;
  featF32[o] = v;
  feat16[o]  = (h16)v;
}

// ---------------------------------------------------------------------------
// Bilinear warp with zero padding outside. Flow read from feat channels.
// ---------------------------------------------------------------------------
__global__ void warp_kernel(const float* __restrict__ img, const float* __restrict__ featF32,
                            int flowCh, float scale, float* __restrict__ out) {
  int idx = blockIdx.x * 256 + threadIdx.x;
  if (idx >= B_ * C_ * HW_) return;
  int b  = idx / (C_ * HW_);
  int rr = idx % (C_ * HW_);
  int c  = rr >> 15;
  int xy = rr & (HW_ - 1);
  int x = xy & (W_ - 1);
  int y = xy >> 8;
  size_t fb = ((size_t)b * CT_ + flowCh) * HW_ + xy;
  float xs = (float)x + featF32[fb] * scale;
  float ys = (float)y + featF32[fb + HW_] * scale;
  float x0 = floorf(xs), y0 = floorf(ys);
  float wx = xs - x0, wy = ys - y0;
  const float* ip = img + ((size_t)b * C_ + c) * HW_;
  float o = 0.f;
  #pragma unroll
  for (int ty = 0; ty < 2; ++ty)
    #pragma unroll
    for (int tx = 0; tx < 2; ++tx) {
      float xi = x0 + tx, yi = y0 + ty;
      float wgt = (tx ? wx : 1.f - wx) * (ty ? wy : 1.f - wy);
      if (xi >= 0.f && xi <= (float)(W_-1) && yi >= 0.f && yi <= (float)(H_-1))
        o += wgt * ip[(int)yi * W_ + (int)xi];
    }
  out[idx] = o;
}

// ---------------------------------------------------------------------------
// Correlation: vol[dy*9+dx] = leaky(mean_c f1 * shift(f2, dy-4, dx-4))
// block = (dy, y, b); threads = x. Channels chunked through LDS.
// ---------------------------------------------------------------------------
__global__ __launch_bounds__(256) void corr_kernel(const float* __restrict__ f1,
    const float* __restrict__ f2, float* __restrict__ featF32, h16* __restrict__ feat16) {
  __shared__ float s1[16 * 256];
  __shared__ float s2[16 * 264];
  int x  = threadIdx.x;
  int dy = blockIdx.x;
  int y  = blockIdx.y;
  int b  = blockIdx.z;
  int ys = y + dy - 4;
  float acc[9];
  #pragma unroll
  for (int i = 0; i < 9; ++i) acc[i] = 0.f;
  for (int c0 = 0; c0 < C_; c0 += 16) {
    __syncthreads();
    for (int t = x; t < 16 * 256; t += 256) {
      int cc = t >> 8, xx = t & 255;
      s1[t] = f1[(((size_t)b * C_ + c0 + cc) * H_ + y) * W_ + xx];
    }
    for (int t = x; t < 16 * 264; t += 256) {
      int cc = t / 264, xx = t % 264;
      int gx = xx - 4;
      float v = 0.f;
      if (ys >= 0 && ys < H_ && gx >= 0 && gx < W_)
        v = f2[(((size_t)b * C_ + c0 + cc) * H_ + ys) * W_ + gx];
      s2[t] = v;
    }
    __syncthreads();
    #pragma unroll 4
    for (int cc = 0; cc < 16; ++cc) {
      float a = s1[cc * 256 + x];
      #pragma unroll
      for (int dx = 0; dx < 9; ++dx)
        acc[dx] += a * s2[cc * 264 + x + dx];
    }
  }
  #pragma unroll
  for (int dx = 0; dx < 9; ++dx) {
    float v = leakyf(acc[dx] * (1.f / 96.f));
    size_t o = (((size_t)b * CT_ + 288 + dy * 9 + dx) * H_ + y) * W_ + x;
    featF32[o] = v;
    feat16[o]  = (h16)v;
  }
}

// ---------------------------------------------------------------------------
// Implicit-GEMM 3x3 conv via v_wmma_f32_16x16x32_f16.
// Workgroup: 8 wave32 = 2 (co tiles of 16) x 4 (pixel tiles of 16) -> 32co x 64px.
// K loop: 32 input channels per chunk staged in LDS (f16), 9 taps = 9 WMMAs.
// A-frag (16x32 f16): lane<16 holds row M=lane, halves K{0..7,16..23};
//                     lane>=16 same rows, K{8..15,24..31}.
// B-frag (32x16 f16): lane<16 holds col N=lane, K{0..7,16..23};
//                     lane>=16 col N=lane-16, K{8..15,24..31}.
// ---------------------------------------------------------------------------
__global__ __launch_bounds__(256) void conv3x3_wmma(
    const h16* __restrict__ feat16, const h16* __restrict__ wrep,
    const float* __restrict__ bias,
    float* __restrict__ outBase, h16* __restrict__ mirror,
    int cinAct, int cinP, int coutAct,
    int chOff, int outChOff, int outChTot, int nCoBlocks, int coutP) {
  __shared__ __align__(16) h16 ldsIn[3 * 66 * 32];   // [row][x][c]
  __shared__ __align__(16) h16 ldsW[9 * 32 * 32];    // [tap][co][k]

  int tid    = threadIdx.x;
  int lane   = tid & 31;
  int wave   = tid >> 5;
  int waveCo = wave >> 2;       // 0..1
  int wavePx = wave & 3;        // 0..3
  int x0  = blockIdx.x * 64;
  int y   = blockIdx.y;
  int b   = blockIdx.z / nCoBlocks;
  int coB = blockIdx.z % nCoBlocks;

  int m      = lane & 15;
  int hiHalf = lane >> 4;
  int kA     = hiHalf * 8;      // A half-K base
  int kB     = hiHalf * 8;      // B K base

  v8f acc = {};

  for (int ci0 = 0; ci0 < cinP; ci0 += 32) {
    __syncthreads();
    // stage activations (x innermost for coalesced global reads)
    for (int t = tid; t < 3 * 66 * 32; t += 256) {
      int xx = t % 66;
      int r3 = (t / 66) % 3;
      int c  = t / 198;
      int gx = x0 - 1 + xx;
      int gy = y - 1 + r3;
      int ch = ci0 + c;
      h16 v = (h16)0.f;
      if (ch < cinAct && (unsigned)gx < (unsigned)W_ && (unsigned)gy < (unsigned)H_)
        v = feat16[(((size_t)b * CT_ + chOff + ch) * H_ + gy) * W_ + gx];
      ldsIn[(r3 * 66 + xx) * 32 + c] = v;
    }
    // stage weights (k innermost, contiguous in repacked layout)
    for (int t = tid; t < 9 * 32 * 32; t += 256) {
      int k   = t & 31;
      int co  = (t >> 5) & 31;
      int tap = t >> 10;
      ldsW[t] = wrep[((size_t)tap * coutP + coB * 32 + co) * cinP + ci0 + k];
    }
    __syncthreads();

    #pragma unroll
    for (int tap = 0; tap < 9; ++tap) {
      int ky = tap / 3, kx = tap % 3;
      int wbase = (tap * 32 + waveCo * 16 + m) * 32;
      union { v16h v; v8h h[2]; } A, Bm;
      A.h[0] = *(const v8h*)&ldsW[wbase + kA];
      A.h[1] = *(const v8h*)&ldsW[wbase + kA + 16];
      int xl = wavePx * 16 + m + kx;
      const h16* bp = &ldsIn[(ky * 66 + xl) * 32 + kB];
      Bm.h[0] = *(const v8h*)bp;
      Bm.h[1] = *(const v8h*)(bp + 16);
      acc = __builtin_amdgcn_wmma_f32_16x16x32_f16(false, A.v, false, Bm.v,
                                                   (short)0, acc, false, false);
    }
  }

  // D layout: VGPR v -> M = v + 8*hiHalf, N = m
  int coBase = coB * 32 + waveCo * 16 + hiHalf * 8;
  int px = x0 + wavePx * 16 + m;
  #pragma unroll
  for (int v = 0; v < 8; ++v) {
    int co = coBase + v;
    if (co < coutAct) {
      float o = leakyf(acc[v] + bias[co]);
      outBase[(((size_t)b * outChTot + outChOff + co) * H_ + y) * W_ + px] = o;
      if (mirror)
        mirror[(((size_t)b * CT_ + outChOff + co) * H_ + y) * W_ + px] = (h16)o;
    }
  }
}

// ---------------------------------------------------------------------------
extern "C" void kernel_launch(void* const* d_in, const int* in_sizes, int n_in,
                              void* d_out, int out_size, void* d_ws, size_t ws_size,
                              hipStream_t stream) {
  (void)in_sizes; (void)n_in; (void)out_size; (void)ws_size;
  const float* tenOne    = (const float*)d_in[0];
  const float* tenTwo    = (const float*)d_in[1];
  const float* prev_flow = (const float*)d_in[2];
  const float* prev_feat = (const float*)d_in[3];
  const float* quat      = (const float*)d_in[4];
  const float* Km        = (const float*)d_in[5];
  const float* Kinv      = (const float*)d_in[6];
  const float* upflow_w  = (const float*)d_in[7];
  const float* upflow_b  = (const float*)d_in[8];
  const float* upfeat_w  = (const float*)d_in[9];
  const float* upfeat_b  = (const float*)d_in[10];
  const float* w1 = (const float*)d_in[11]; const float* b1 = (const float*)d_in[12];
  const float* w2 = (const float*)d_in[13]; const float* b2 = (const float*)d_in[14];
  const float* w3 = (const float*)d_in[15]; const float* b3 = (const float*)d_in[16];
  const float* w4 = (const float*)d_in[17]; const float* b4 = (const float*)d_in[18];

  float* flowOut = (float*)d_out;                       // (B,2,H,W)
  float* featF32 = flowOut + (size_t)B_ * 2 * HW_;      // (B,471,H,W)

  // workspace carve
  char* ws = (char*)d_ws;
  size_t off = 0;
  auto carve = [&](size_t bytes) -> void* {
    void* p = ws + off;
    off = (off + bytes + 255) & ~(size_t)255;
    return p;
  };
  h16*   feat16 = (h16*)  carve((size_t)B_ * CT_ * HW_ * sizeof(h16));
  float* t2w    = (float*)carve((size_t)B_ * C_ * HW_ * sizeof(float));
  float* t2ww   = (float*)carve((size_t)B_ * C_ * HW_ * sizeof(float));
  h16*   wr1    = (h16*)  carve((size_t)9 * 128 * 192 * sizeof(h16));
  h16*   wr2    = (h16*)  carve((size_t)9 * 96  * 320 * sizeof(h16));
  h16*   wr3    = (h16*)  carve((size_t)9 * 64  * 416 * sizeof(h16));
  h16*   wr4    = (h16*)  carve((size_t)9 * 32  * 480 * sizeof(h16));

  dim3 blk(256);
  auto g1 = [](int n) { return dim3((unsigned)((n + 255) / 256)); };

  // weight repack (independent)
  repack_kernel<<<g1(9*128*192), blk, 0, stream>>>(w1, wr1, 128, 183, 128, 192);
  repack_kernel<<<g1(9*96*320),  blk, 0, stream>>>(w2, wr2,  96, 311,  96, 320);
  repack_kernel<<<g1(9*64*416),  blk, 0, stream>>>(w3, wr3,  64, 407,  64, 416);
  repack_kernel<<<g1(9*32*480),  blk, 0, stream>>>(w4, wr4,   2, 471,  32, 480);

  // upsampled flow/feat -> feat ch 467-468 / 469-470
  upconv_kernel<<<g1(B_*HW_), blk, 0, stream>>>(prev_flow, upflow_w, upflow_b,   2, 467, featF32, feat16);
  upconv_kernel<<<g1(B_*HW_), blk, 0, stream>>>(prev_feat, upfeat_w, upfeat_b, 503, 469, featF32, feat16);
  // rotation flow -> ch 369-370
  rflow_kernel<<<g1(B_*HW_), blk, 0, stream>>>(quat, Km, Kinv, featF32, feat16);
  // tenOne -> ch 371-466
  copy_tenone_kernel<<<g1(B_*C_*HW_), blk, 0, stream>>>(tenOne, featF32, feat16);
  // warp(tenTwo, Rflow) then warp(., 2*tenFlow)
  warp_kernel<<<g1(B_*C_*HW_), blk, 0, stream>>>(tenTwo, featF32, 369, 1.0f, t2w);
  warp_kernel<<<g1(B_*C_*HW_), blk, 0, stream>>>(t2w,    featF32, 467, 2.0f, t2ww);
  // correlation volume -> ch 288-368
  corr_kernel<<<dim3(9, H_, B_), blk, 0, stream>>>(tenOne, t2ww, featF32, feat16);

  // dense-net conv chain (WMMA); each writes leaky output in front of its input range
  conv3x3_wmma<<<dim3(W_/64, H_, B_*4), blk, 0, stream>>>(
      feat16, wr1, b1, featF32, feat16, 183, 192, 128, 288, 160, CT_, 4, 128);
  conv3x3_wmma<<<dim3(W_/64, H_, B_*3), blk, 0, stream>>>(
      feat16, wr2, b2, featF32, feat16, 311, 320,  96, 160,  64, CT_, 3,  96);
  conv3x3_wmma<<<dim3(W_/64, H_, B_*2), blk, 0, stream>>>(
      feat16, wr3, b3, featF32, feat16, 407, 416,  64,  64,   0, CT_, 2,  64);
  conv3x3_wmma<<<dim3(W_/64, H_, B_*1), blk, 0, stream>>>(
      feat16, wr4, b4, flowOut, (h16*)nullptr, 471, 480, 2, 0, 0, 2, 1, 32);
}
